// SNNNetwork_7971459301376
// MI455X (gfx1250) — compile-verified, matrix operand order checked by reading, not run on hardware
//
#include <hip/hip_runtime.h>

typedef float    v2f  __attribute__((ext_vector_type(2)));
typedef float    v4f  __attribute__((ext_vector_type(4)));
typedef float    v8f  __attribute__((ext_vector_type(8)));
typedef _Float16 v8h  __attribute__((ext_vector_type(8)));
typedef _Float16 v16h __attribute__((ext_vector_type(16)));

#define NB 4096
#define NT 256
#define NI 7
#define NH 64
#define NO 8
#define BETA   0.8f
#define THRESH 1.0f

// padded row stride (in halves) for the spike tile -> conflict-free ds_load_b128
#define SPK_STRIDE 72

#if __has_builtin(__builtin_amdgcn_wmma_f32_16x16x4_f32)
#define HAVE_WMMA_F32X4 1
#else
#define HAVE_WMMA_F32X4 0
#endif

// One wave32 per 16-batch tile. Layouts (per CDNA5 ISA 7.12.2):
//   C/D 16x16 f32 : vgpr r, lane l -> (M = r + 8*(l>>4), N = l&15)
//   A   16x4  f32 : lane l -> M = l&15 ; v0=K(l<16?0:2), v1=K+1
//   A   16x32 f16 : lane l -> M = l&15 ; k = {0..7,16..23} (+8 for upper half)
//   B mirrors A with M<->N.
__launch_bounds__(32)
__global__ void snn_fused_wmma(const float* __restrict__ x,
                               const float* __restrict__ hidden0,
                               const float* __restrict__ W1,
                               const float* __restrict__ b1,
                               const float* __restrict__ W2,
                               const float* __restrict__ b2,
                               float* __restrict__ out_y,    // (B,T,O)
                               float* __restrict__ out_mem)  // (B,T,H)
{
    __shared__ __align__(16) float    memLds[16 * NH];         // 4 KB
    __shared__ __align__(16) _Float16 spkLds[16 * SPK_STRIDE]; // 2.25 KB
    __shared__ __align__(16) float    outLds[16 * NO];         // 0.5 KB

    const int lane = threadIdx.x;   // 0..31
    const int hf   = lane >> 4;     // lane half
    const int lm   = lane & 15;
    const int b0   = blockIdx.x * 16;

    // ---------------- constant fragments ----------------
#if HAVE_WMMA_F32X4
    // W1^T as f32 B fragments: [k-chunk][h-tile], 2 VGPRs each.
    // Chunk1 row K=7 (lanes>=16, .y) is zeroed: it absorbs the NI=7 pad, so
    // the matching A slot may hold any finite value.
    v2f w1B[2][4];
#pragma unroll
    for (int c = 0; c < 2; ++c)
#pragma unroll
        for (int ht = 0; ht < 4; ++ht) {
            int h  = ht * 16 + lm;
            int kb = c * 4 + hf * 2;
            w1B[c][ht].x = (kb     < NI) ? W1[h * NI + kb]     : 0.0f;
            w1B[c][ht].y = (kb + 1 < NI) ? W1[h * NI + kb + 1] : 0.0f;
        }
    // bias b1 pre-loaded into the accumulator (same h for every row M)
    v8f c1bias[4];
#pragma unroll
    for (int ht = 0; ht < 4; ++ht) {
        float bv = b1[ht * 16 + lm];
#pragma unroll
        for (int e = 0; e < 8; ++e) c1bias[ht][e] = bv;
    }
#else
    // scalar fallback: per-lane private copies of W1 rows
    float w1loc[4][NI];
    float b1loc[4];
#pragma unroll
    for (int ht = 0; ht < 4; ++ht) {
        int h = ht * 16 + lm;
        b1loc[ht] = b1[h];
#pragma unroll
        for (int k = 0; k < NI; ++k) w1loc[ht][k] = W1[h * NI + k];
    }
#endif

    // W2^T as f16 B fragments (spikes are exactly 0/1 in f16): [k-chunk]
    v16h w2B[2];
    {
        int o = lm;
#pragma unroll
        for (int c = 0; c < 2; ++c)
#pragma unroll
            for (int v = 0; v < 8; ++v) {
                int k = c * 32 + (v < 4 ? 2 * v : 16 + 2 * (v - 4)) + hf * 8;
                _Float16 e0 = (_Float16)0.0f, e1 = (_Float16)0.0f;
                if (o < NO) {
                    e0 = (_Float16)W2[o * NH + k];
                    e1 = (_Float16)W2[o * NH + k + 1];
                }
                w2B[c][2 * v]     = e0;
                w2B[c][2 * v + 1] = e1;
            }
    }
    v8f c2bias;
    {
        float bv = (lm < NO) ? b2[lm] : 0.0f;
#pragma unroll
        for (int e = 0; e < 8; ++e) c2bias[e] = bv;
    }

    // ---------------- persistent LIF state (C layout) ----------------
    v8f mem[4];
#pragma unroll
    for (int ht = 0; ht < 4; ++ht)
#pragma unroll
        for (int r = 0; r < 8; ++r) {
            int bl = r + 8 * hf;
            mem[ht][r] = hidden0[(b0 + bl) * NH + ht * 16 + lm];
        }

    const int k0   = hf * 2;          // chunk0 A columns: {0,1} / {2,3}
    const int xrow = b0 + lm;         // A-matrix row (same for both halves)

    for (int t = 0; t < NT; ++t) {
        // ------- matmul1: cur = x_t * W1^T + b1 -------
        v8f cur[4];
#if HAVE_WMMA_F32X4
        {
            const int base = (xrow * NT + t) * NI;
            v2f a0, a1;
            a0.x = x[base + k0];
            a0.y = x[base + k0 + 1];
            a1.x = x[base + 4 + 2 * hf];   // k=4 / k=6 (always in-bounds)
            a1.y = x[base + 5];            // k=5 / k=7-slot (x7 pad handled by B=0)
#pragma unroll
            for (int ht = 0; ht < 4; ++ht) {
                v8f c0 = __builtin_amdgcn_wmma_f32_16x16x4_f32(
                    false, a0, false, w1B[0][ht], (short)0, c1bias[ht], false, false);
                cur[ht] = __builtin_amdgcn_wmma_f32_16x16x4_f32(
                    false, a1, false, w1B[1][ht], (short)0, c0, false, false);
            }
        }
#else
#pragma unroll
        for (int r = 0; r < 8; ++r) {
            int bl = r + 8 * hf;
            const float* xr = x + ((b0 + bl) * NT + t) * NI;
            float xv[NI];
#pragma unroll
            for (int k = 0; k < NI; ++k) xv[k] = xr[k];
#pragma unroll
            for (int ht = 0; ht < 4; ++ht) {
                float a = b1loc[ht];
#pragma unroll
                for (int k = 0; k < NI; ++k) a = fmaf(xv[k], w1loc[ht][k], a);
                cur[ht][r] = a;
            }
        }
#endif

        // ------- LIF update + stage spk (f16) and mem (f32) in LDS -------
#pragma unroll
        for (int ht = 0; ht < 4; ++ht)
#pragma unroll
            for (int r = 0; r < 8; ++r) {
                float m     = mem[ht][r];
                float reset = (m > THRESH) ? THRESH : 0.0f;
                m = BETA * m + cur[ht][r] - reset;
                _Float16 spk = (m > THRESH) ? (_Float16)1.0f : (_Float16)0.0f;
                mem[ht][r] = m;
                int bl = r + 8 * hf;
                int h  = ht * 16 + lm;
                memLds[bl * NH + h]         = m;
                spkLds[bl * SPK_STRIDE + h] = spk;
            }
        __syncthreads();   // single-wave WG: HW NOP, compiler fence

        // ------- matmul2: y_t = spk * W2^T + b2 (f16 WMMA, K=64) -------
        v16h a2[2];
        {
            const _Float16* row = &spkLds[lm * SPK_STRIDE];
#pragma unroll
            for (int c = 0; c < 2; ++c) {
                int kb = c * 32 + hf * 8;
                v8h lo = *(const v8h*)(row + kb);
                v8h hi = *(const v8h*)(row + kb + 16);
                a2[c] = __builtin_shufflevector(lo, hi,
                        0, 1, 2, 3, 4, 5, 6, 7, 8, 9, 10, 11, 12, 13, 14, 15);
            }
        }
        v8f d2 = __builtin_amdgcn_wmma_f32_16x16x32_f16(
            false, a2[0], false, w2B[0], (short)0, c2bias, false, false);
        d2 = __builtin_amdgcn_wmma_f32_16x16x32_f16(
            false, a2[1], false, w2B[1], (short)0, d2, false, false);

        // read back mem tile rows (coalesced source for global stores)
        v4f mrow[8];
#pragma unroll
        for (int s = 0; s < 8; ++s) {
            int row = s * 2 + hf;
            mrow[s] = *(const v4f*)&memLds[row * NH + lm * 4];
        }

        // stage output tile (16b x 8o) in LDS for one coalesced store
        if (lm < NO) {
#pragma unroll
            for (int r = 0; r < 8; ++r) {
                int bl = r + 8 * hf;
                outLds[bl * NO + lm] = d2[r];
            }
        }
        __syncthreads();

        // ------- coalesced global stores -------
        // mems: 256 B per batch row
#pragma unroll
        for (int s = 0; s < 8; ++s) {
            int row = s * 2 + hf;
            *(v4f*)&out_mem[((b0 + row) * NT + t) * NH + lm * 4] = mrow[s];
        }
        // outputs: whole 512 B tile in one b128 per lane
        {
            v4f yv = *(const v4f*)&outLds[lane * 4];
            int bl = lane >> 1;
            int oc = (lane & 1) * 4;
            *(v4f*)&out_y[((b0 + bl) * NT + t) * NO + oc] = yv;
        }
        __syncthreads();
    }
}

extern "C" void kernel_launch(void* const* d_in, const int* in_sizes, int n_in,
                              void* d_out, int out_size, void* d_ws, size_t ws_size,
                              hipStream_t stream) {
    const float* x  = (const float*)d_in[0];
    const float* h0 = (const float*)d_in[1];
    const float* W1 = (const float*)d_in[2];
    const float* b1 = (const float*)d_in[3];
    const float* W2 = (const float*)d_in[4];
    const float* b2 = (const float*)d_in[5];

    float* out      = (float*)d_out;
    float* out_y    = out;                         // 4096*256*8
    float* out_mem  = out + (size_t)NB * NT * NO;  // 4096*256*64

    dim3 grid(NB / 16);
    dim3 block(32);
    hipLaunchKernelGGL(snn_fused_wmma, grid, block, 0, stream,
                       x, h0, W1, b1, W2, b2, out_y, out_mem);
}
